// MultiHeadAttention_23759759082068
// MI455X (gfx1250) — matrix operand order repaired and checked by reading, not compiled, not run on hardware
//
#include <hip/hip_runtime.h>
#include <hip/hip_bf16.h>

// MHA: B=2, S=2048, D=1024, H=16, HD=64. bf16 WMMA (16x16x32), fp32 softmax.
// v2: attention shares K/V tiles across the workgroup via async-to-LDS
// (double buffered, ASYNCcnt), GEMM upsized to 64x32 per wave.

#define BB  2
#define SS  2048
#define DD  1024
#define HH  16
#define HDD 64
#define MM  (BB*SS)   // 4096

typedef __attribute__((ext_vector_type(16))) __bf16 v16bf;
typedef __attribute__((ext_vector_type(8)))  float  v8f;

union Frag16 { v16bf v; uint4 q[2]; };

// ---- WMMA fragment loaders (wave32 layouts per CDNA5 ISA 7.12.2) ----------

// A 16x32 (MxK) from row-major src, leading dim ld (elements).
static __device__ __forceinline__ v16bf load_a_frag(const __bf16* src, int ld) {
  const int lane = threadIdx.x & 31;
  const int g = lane >> 4, m = lane & 15;
  const __bf16* p = src + (size_t)m * ld + g * 8;
  Frag16 f;
  f.q[0] = *(const uint4*)(p);        // K = g*8 + 0..7
  f.q[1] = *(const uint4*)(p + 16);   // K = 16 + g*8 + 0..7
  return f.v;
}

// B 32x16 (KxN) where B[k,n] = src[n*ld + k] (NT access): K = g*8+0..7, 16+g*8+0..7.
static __device__ __forceinline__ v16bf load_bT_frag(const __bf16* src, int ld) {
  const int lane = threadIdx.x & 31;
  const int g = lane >> 4, n = lane & 15;
  const __bf16* p = src + (size_t)n * ld + g * 8;
  Frag16 f;
  f.q[0] = *(const uint4*)(p);
  f.q[1] = *(const uint4*)(p + 16);
  return f.v;
}

static __device__ __forceinline__ v8f wmma_bf16(v16bf a, v16bf b, v8f c) {
  return __builtin_amdgcn_wmma_f32_16x16x32_bf16(
      false, a, false, b, (short)0, c, false, false);
}

// CDNA5 async copy: per-lane 16B global -> LDS, tracked by ASYNCcnt.
static __device__ __forceinline__ void async_copy_b128(unsigned lds_off,
                                                       const __bf16* gptr) {
  asm volatile("global_load_async_to_lds_b128 %0, %1, off"
               :: "v"(lds_off), "v"(gptr) : "memory");
}
static __device__ __forceinline__ void wait_async0() {
  asm volatile("s_wait_asynccnt 0" ::: "memory");
}
static __device__ __forceinline__ void wait_ds0() {
  asm volatile("s_wait_dscnt 0" ::: "memory");
}

static __device__ __forceinline__ float rowmax16(float v) {
#pragma unroll
  for (int off = 8; off >= 1; off >>= 1) v = fmaxf(v, __shfl_xor(v, off, 16));
  return v;
}
static __device__ __forceinline__ float rowsum16(float v) {
#pragma unroll
  for (int off = 8; off >= 1; off >>= 1) v += __shfl_xor(v, off, 16);
  return v;
}

// ---- Stage 0: fp32 -> bf16 -------------------------------------------------
__global__ void cvt_f32_bf16(const float* __restrict__ in,
                             __bf16* __restrict__ out, int n) {
  int i = blockIdx.x * blockDim.x + threadIdx.x;
  if (i < n) out[i] = (__bf16)in[i];
}

// ---- Stages 1 & 3: NT-GEMM  y = A(MxK) * W(NxK)^T --------------------------
// 8 waves = 2(m) x 4(n); wave tile 64x32 (8 WMMAs per 6 frag loads); WG 128x128.
// mode 0: Q->bf16 [B,H,S,HD] *1/8   mode 1: K->bf16 [B,H,S,HD]
// mode 2: V->bf16 [B,H,HD,S]        mode 3: fp32 out = c + bias
__global__ __launch_bounds__(256)
void gemm_nt_bf16(const __bf16* __restrict__ A, const __bf16* __restrict__ W,
                  __bf16* __restrict__ obf, float* __restrict__ ofp,
                  const float* __restrict__ bias, int mode) {
  const int lane = threadIdx.x & 31;
  const int wave = threadIdx.x >> 5;
  const int wm = wave >> 2, wn = wave & 3;
  const int m0 = blockIdx.y * 128 + wm * 64;
  const int n0 = blockIdx.x * 128 + wn * 32;

  v8f c[4][2] = {};
  for (int kk = 0; kk < DD; kk += 32) {
    // prefetch one k-tile ahead (global_prefetch_b8)
    __builtin_prefetch(A + (size_t)m0 * DD + kk + 32, 0, 1);
    __builtin_prefetch(W + (size_t)n0 * DD + kk + 32, 0, 1);
    v16bf b0 = load_bT_frag(W + (size_t)n0 * DD + kk, DD);
    v16bf b1 = load_bT_frag(W + (size_t)(n0 + 16) * DD + kk, DD);
#pragma unroll
    for (int i = 0; i < 4; ++i) {
      v16bf ai = load_a_frag(A + (size_t)(m0 + i * 16) * DD + kk, DD);
      c[i][0] = wmma_bf16(ai, b0, c[i][0]);
      c[i][1] = wmma_bf16(ai, b1, c[i][1]);
    }
  }

  const int g = lane >> 4, nl = lane & 15;
#pragma unroll
  for (int i = 0; i < 4; ++i)
#pragma unroll
    for (int j = 0; j < 2; ++j)
#pragma unroll
      for (int r = 0; r < 8; ++r) {
        int m = m0 + i * 16 + r + 8 * g;
        int n = n0 + j * 16 + nl;
        float val = c[i][j][r];
        if (mode == 3) {
          ofp[(size_t)m * DD + n] = val + bias[n];
        } else {
          int b = m / SS, s = m % SS, h = n / HDD, hd = n % HDD;
          if (mode == 2) {
            obf[(((size_t)b * HH + h) * HDD + hd) * SS + s] = (__bf16)val;
          } else {
            float sc = (mode == 0) ? 0.125f : 1.0f;
            obf[(((size_t)b * HH + h) * SS + s) * HDD + hd] = (__bf16)(val * sc);
          }
        }
      }
}

// ---- Stage 2: flash attention, WG-shared K/V tiles via async-to-LDS --------
// Block = 8 waves, one (b,h), 128 consecutive queries (16 per wave).
// Key tiles of 32 streamed through double-buffered LDS.
__global__ __launch_bounds__(256)
void attn_fwd(const __bf16* __restrict__ Q, const __bf16* __restrict__ K,
              const __bf16* __restrict__ Vt, __bf16* __restrict__ C) {
  __shared__ __bf16 kbuf[2][32 * 64];   // [key, hd]   4KB each
  __shared__ __bf16 vbuf[2][64 * 32];   // [hd, key]   4KB each
  __shared__ __bf16 pbufs[8][16 * 32];  // per-wave P tile, 1KB each

  const int tid  = threadIdx.x;
  const int lane = tid & 31;
  const int wave = tid >> 5;
  const int qb = blockIdx.x & 15;            // 16 q-blocks of 128
  const int h  = (blockIdx.x >> 4) & 15;
  const int b  = blockIdx.x >> 8;

  const __bf16* Qh = Q  + (((size_t)b * HH + h) * SS + qb * 128 + wave * 16) * HDD;
  const __bf16* Kh = K  + ((size_t)b * HH + h) * SS * HDD;
  const __bf16* Vh = Vt + ((size_t)b * HH + h) * HDD * SS;

  // cooperative tile staging: 256 threads x 16B cover each 4KB tile
  const int krow = tid >> 3, kchunk = (tid & 7) * 8;   // K: 32 rows x 64
  const int vrow = tid >> 2, vchunk = (tid & 3) * 8;   // V: 64 rows x 32
  auto stage = [&](int kt, int buf) {
    async_copy_b128((unsigned)(uintptr_t)&kbuf[buf][krow * 64 + kchunk],
                    Kh + (size_t)(kt * 32 + krow) * HDD + kchunk);
    async_copy_b128((unsigned)(uintptr_t)&vbuf[buf][vrow * 32 + vchunk],
                    Vh + (size_t)vrow * SS + kt * 32 + vchunk);
  };

  v16bf aq0 = load_a_frag(Qh, HDD);        // Q pre-scaled by 1/sqrt(HD)
  v16bf aq1 = load_a_frag(Qh + 32, HDD);

  v8f ctx[4] = {};
  float rmax[8], rsum[8];
#pragma unroll
  for (int r = 0; r < 8; ++r) { rmax[r] = -INFINITY; rsum[r] = 0.0f; }

  const int g = lane >> 4, nl = lane & 15;
  __bf16* pbuf = &pbufs[wave][0];

  stage(0, 0);
  const int NT = SS / 32;                  // 64 key tiles
  for (int kt = 0; kt < NT; ++kt) {
    const int cur = kt & 1;
    wait_async0();                         // my share of tile kt landed
    __syncthreads();                       // everyone's share landed
    if (kt + 1 < NT) stage(kt + 1, cur ^ 1);

    const __bf16* kb = &kbuf[cur][0];
    const __bf16* vb = &vbuf[cur][0];

    // ---- scores: 16x32 tile = two 16x16 C frags (K contraction = HD = 64)
    v8f s0 = {}, s1 = {};
    {
      v16bf bk = load_bT_frag(kb, 64);
      s0 = wmma_bf16(aq0, bk, s0);
      bk = load_bT_frag(kb + 32, 64);
      s0 = wmma_bf16(aq1, bk, s0);
      bk = load_bT_frag(kb + 16 * 64, 64);
      s1 = wmma_bf16(aq0, bk, s1);
      bk = load_bT_frag(kb + 16 * 64 + 32, 64);
      s1 = wmma_bf16(aq1, bk, s1);
    }

    // ---- online softmax (row = r + 8*lanegroup)
#pragma unroll
    for (int r = 0; r < 8; ++r) {
      float tmax = rowmax16(fmaxf(s0[r], s1[r]));
      float nm   = fmaxf(rmax[r], tmax);
      float corr = __expf(rmax[r] - nm);
      rmax[r] = nm;
      float p0 = __expf(s0[r] - nm);
      float p1 = __expf(s1[r] - nm);
      rsum[r] = rsum[r] * corr + rowsum16(p0 + p1);
#pragma unroll
      for (int j = 0; j < 4; ++j) ctx[j][r] *= corr;
      int m = r + 8 * g;
      pbuf[m * 32 + nl]      = (__bf16)p0;
      pbuf[m * 32 + 16 + nl] = (__bf16)p1;
    }

    wait_ds0();                            // P stores -> A-frag reload (same wave)
    v16bf ap = load_a_frag(pbuf, 32);
#pragma unroll
    for (int j = 0; j < 4; ++j) {
      v16bf bv = load_bT_frag(vb + j * 16 * 32, 32);
      ctx[j] = wmma_bf16(ap, bv, ctx[j]);
    }
  }

  // ---- normalize, write concat-heads bf16 [B*S, D]
#pragma unroll
  for (int r = 0; r < 8; ++r) {
    float inv = 1.0f / rsum[r];
    int m = b * SS + qb * 128 + wave * 16 + r + 8 * g;
#pragma unroll
    for (int j = 0; j < 4; ++j) {
      int col = h * HDD + j * 16 + nl;
      C[(size_t)m * DD + col] = (__bf16)(ctx[j][r] * inv);
    }
  }
}

// ---- host-side orchestration ----------------------------------------------
extern "C" void kernel_launch(void* const* d_in, const int* in_sizes, int n_in,
                              void* d_out, int out_size, void* d_ws, size_t ws_size,
                              hipStream_t stream) {
  (void)in_sizes; (void)n_in; (void)out_size; (void)ws_size;
  const float* x  = (const float*)d_in[0];
  const float* wq = (const float*)d_in[1];
  const float* wk = (const float*)d_in[2];
  const float* wv = (const float*)d_in[3];
  const float* wo = (const float*)d_in[4];
  const float* bo = (const float*)d_in[5];
  float* out = (float*)d_out;

  char* ws = (char*)d_ws;
  const size_t MB = 1024 * 1024;
  __bf16* xb  = (__bf16*)(ws);             //  8 MB  x bf16 [4096,1024]
  __bf16* wqb = (__bf16*)(ws +  8 * MB);
  __bf16* wkb = (__bf16*)(ws + 10 * MB);
  __bf16* wvb = (__bf16*)(ws + 12 * MB);
  __bf16* wob = (__bf16*)(ws + 14 * MB);
  __bf16* Qb  = (__bf16*)(ws + 16 * MB);   //  8 MB [B,H,S,HD] (pre-scaled)
  __bf16* Kb  = (__bf16*)(ws + 24 * MB);   //  8 MB [B,H,S,HD]
  __bf16* Vtb = (__bf16*)(ws + 32 * MB);   //  8 MB [B,H,HD,S]
  __bf16* Cb  = (__bf16*)(ws + 40 * MB);   //  8 MB concat heads [4096,1024]

  const int nx = MM * DD, nw = DD * DD;
  cvt_f32_bf16<<<(nx + 255) / 256, 256, 0, stream>>>(x,  xb,  nx);
  cvt_f32_bf16<<<(nw + 255) / 256, 256, 0, stream>>>(wq, wqb, nw);
  cvt_f32_bf16<<<(nw + 255) / 256, 256, 0, stream>>>(wk, wkb, nw);
  cvt_f32_bf16<<<(nw + 255) / 256, 256, 0, stream>>>(wv, wvb, nw);
  cvt_f32_bf16<<<(nw + 255) / 256, 256, 0, stream>>>(wo, wob, nw);

  dim3 grid(DD / 128, MM / 128);   // (8, 32)
  gemm_nt_bf16<<<grid, 256, 0, stream>>>(xb, wqb, Qb,  nullptr, nullptr, 0);
  gemm_nt_bf16<<<grid, 256, 0, stream>>>(xb, wkb, Kb,  nullptr, nullptr, 1);
  gemm_nt_bf16<<<grid, 256, 0, stream>>>(xb, wvb, Vtb, nullptr, nullptr, 2);

  attn_fwd<<<BB * HH * 16, 256, 0, stream>>>(Qb, Kb, Vtb, Cb);

  gemm_nt_bf16<<<grid, 256, 0, stream>>>(Cb, wob, nullptr, out, bo, 3);
}
